// RSSMWorldModel_54924041781661
// MI455X (gfx1250) — matrix operand order repaired and checked by reading, not compiled, not run on hardware
//
#include <hip/hip_runtime.h>
#include <hip/hip_bf16.h>

// ---------------------------------------------------------------------------
// RSSM world-model loss on MI455X (gfx1250): bf16 WMMA GEMMs + f32 state.
// GEMMs use 2x2 register-blocked wave tiles (32x32 C per wave, 4 WMMA/K-step).
// ---------------------------------------------------------------------------

#define HID  512
#define LAT  128
#define OBSD 768
#define AEMB 64
#define BB   256
#define SS   128
#define GIN  192          // LAT + AEMB
#define G3   1536         // 3*HID
#define PIN  1024         // 2*HID
#define HZD  640          // HID + LAT

typedef __bf16 bf16;
typedef __attribute__((ext_vector_type(16))) __bf16 v16bf;
typedef __attribute__((ext_vector_type(8)))  __bf16 v8bf;
typedef __attribute__((ext_vector_type(8)))  float  v8f;

__device__ __forceinline__ float sigf(float x) { return 1.0f / (1.0f + __expf(-x)); }

// ---------------------------------------------------------------------------
// WMMA GEMM:  C(MxN, f32 acc) = A(MxK) * Bt(NxK)^T + bias
//   EPI = 0 : store f32
//   EPI = 1 : ELU -> store bf16
//   EPI = 2 : clip[-10,2] -> store f32
//   AF32    : A operand is f32 in memory (converted to bf16 in regs)
// One wave computes a 32x32 C block = 2x2 grid of 16x16 WMMA tiles.
// M,N multiples of 32; K multiple of 32.
// ---------------------------------------------------------------------------
template<int EPI, bool AF32>
__global__ void gemm_wmma(const void* __restrict__ Ap, const bf16* __restrict__ Bt,
                          const float* __restrict__ bias, void* __restrict__ Cp,
                          int M, int N, int K)
{
    const int blkM = M >> 5;
    const int blkN = N >> 5;
    const int wid = blockIdx.x * (blockDim.x >> 5) + ((int)threadIdx.x >> 5);
    if (wid >= blkM * blkN) return;               // wave-uniform: EXEC stays full
    const int tm2  = wid % blkM;
    const int tn2  = wid / blkM;
    const int lane = (int)threadIdx.x & 31;
    const int half = lane >> 4;
    const int lc   = lane & 15;

    // A: lanes 0-15 and 16-31 both hold rows M=0..15 (different K halves)
    const int rowA0 = tm2 * 32 + lc;
    const int rowB0 = tn2 * 32 + lc;              // Bt row == output column
    const bf16* bRow0 = Bt + (size_t)rowB0 * K;
    const bf16* bRow1 = Bt + (size_t)(rowB0 + 16) * K;

    v8f acc[2][2];
    {
        const float bv0 = bias[tn2 * 32 + lc];
        const float bv1 = bias[tn2 * 32 + 16 + lc];
        #pragma unroll
        for (int v = 0; v < 8; ++v) {
            acc[0][0][v] = bv0; acc[1][0][v] = bv0;
            acc[0][1][v] = bv1; acc[1][1][v] = bv1;
        }
    }

    for (int k0 = 0; k0 < K; k0 += 32) {
        v16bf av[2], bv[2];
        if (AF32) {
            #pragma unroll
            for (int i = 0; i < 2; ++i) {
                const float* aRow = (const float*)Ap + (size_t)(rowA0 + 16 * i) * K;
                v8f a0 = *(const v8f*)(aRow + k0 + 8 * half);
                v8f a1 = *(const v8f*)(aRow + k0 + 16 + 8 * half);
                #pragma unroll
                for (int e = 0; e < 8; ++e) { av[i][e] = (bf16)a0[e]; av[i][8 + e] = (bf16)a1[e]; }
            }
        } else {
            #pragma unroll
            for (int i = 0; i < 2; ++i) {
                const bf16* aRow = (const bf16*)Ap + (size_t)(rowA0 + 16 * i) * K;
                v8bf a0 = *(const v8bf*)(aRow + k0 + 8 * half);
                v8bf a1 = *(const v8bf*)(aRow + k0 + 16 + 8 * half);
                av[i] = __builtin_shufflevector(a0, a1, 0,1,2,3,4,5,6,7,8,9,10,11,12,13,14,15);
            }
        }
        {
            v8bf b0 = *(const v8bf*)(bRow0 + k0 + 8 * half);
            v8bf b1 = *(const v8bf*)(bRow0 + k0 + 16 + 8 * half);
            bv[0] = __builtin_shufflevector(b0, b1, 0,1,2,3,4,5,6,7,8,9,10,11,12,13,14,15);
            v8bf b2 = *(const v8bf*)(bRow1 + k0 + 8 * half);
            v8bf b3 = *(const v8bf*)(bRow1 + k0 + 16 + 8 * half);
            bv[1] = __builtin_shufflevector(b2, b3, 0,1,2,3,4,5,6,7,8,9,10,11,12,13,14,15);
        }
        if (k0 + 32 < K) {                         // global_prefetch_b8 next B lines
            __builtin_prefetch(bRow0 + k0 + 32, 0, 1);
            __builtin_prefetch(bRow1 + k0 + 32, 0, 1);
        }
        #pragma unroll
        for (int i = 0; i < 2; ++i)
            #pragma unroll
            for (int j = 0; j < 2; ++j)
                acc[i][j] = __builtin_amdgcn_wmma_f32_16x16x32_bf16(
                                false, av[i], false, bv[j], (short)0, acc[i][j], false, false);
    }

    #pragma unroll
    for (int i = 0; i < 2; ++i) {
        #pragma unroll
        for (int j = 0; j < 2; ++j) {
            const int colC = tn2 * 32 + j * 16 + lc;
            #pragma unroll
            for (int v = 0; v < 8; ++v) {
                const int rowC = tm2 * 32 + i * 16 + 8 * half + v;  // D: M = v + 8*half
                float x = acc[i][j][v];
                if (EPI == 0) {
                    ((float*)Cp)[(size_t)rowC * N + colC] = x;
                } else if (EPI == 1) {
                    x = x > 0.0f ? x : (__expf(x) - 1.0f);
                    ((bf16*)Cp)[(size_t)rowC * N + colC] = (bf16)x;
                } else {
                    ((float*)Cp)[(size_t)rowC * N + colC] = fminf(2.0f, fmaxf(-10.0f, x));
                }
            }
        }
    }
}

// ------------------------- elementwise / setup kernels ---------------------

__global__ void k_init(float* h, bf16* hb, bf16* zb, float* acc)
{
    int idx = blockIdx.x * blockDim.x + threadIdx.x;
    if (idx < BB * HID) { h[idx] = 0.0f; hb[idx] = (bf16)0.0f; }
    if (idx < BB * LAT) zb[idx] = (bf16)0.0f;
    if (idx < 4) acc[idx] = 0.0f;
}

// Wt[n*K + k] = (bf16) W[k*N + n]   (store transposed so B fetches are contiguous)
__global__ void k_transpose(const float* __restrict__ W, bf16* __restrict__ Wt, int K, int N)
{
    int idx = blockIdx.x * blockDim.x + threadIdx.x;
    if (idx >= K * N) return;
    int k = idx / N, n = idx - k * N;
    Wt[(size_t)n * K + k] = (bf16)W[idx];
}

__global__ void k_aemb(const float* __restrict__ tab, const int* __restrict__ act,
                       bf16* __restrict__ out)
{
    int idx = blockIdx.x * blockDim.x + threadIdx.x;
    if (idx >= BB * SS * AEMB) return;
    int bs = idx / AEMB, e = idx - bs * AEMB;
    out[idx] = (bf16)tab[(size_t)act[bs] * AEMB + e];
}

__global__ void k_build_x(const bf16* __restrict__ zb, const bf16* __restrict__ aemb,
                          bf16* __restrict__ x, int t)
{
    int idx = blockIdx.x * blockDim.x + threadIdx.x;
    if (idx >= BB * GIN) return;
    int b = idx / GIN, j = idx - b * GIN;
    x[idx] = (j < LAT) ? zb[b * LAT + j]
                       : aemb[((size_t)(b * SS + t)) * AEMB + (j - LAT)];
}

__global__ void k_gru(const float* __restrict__ gx, const float* __restrict__ gh,
                      float* __restrict__ h, bf16* __restrict__ hb)
{
    int idx = blockIdx.x * blockDim.x + threadIdx.x;
    if (idx >= BB * HID) return;
    int b = idx / HID, j = idx - b * HID;
    size_t base = (size_t)b * G3;
    float r = sigf(gx[base + j]           + gh[base + j]);
    float u = sigf(gx[base + HID + j]     + gh[base + HID + j]);
    float n = tanhf(gx[base + 2*HID + j]  + r * gh[base + 2*HID + j]);
    float hn = (1.0f - u) * n + u * h[idx];
    h[idx]  = hn;
    hb[idx] = (bf16)hn;
}

__global__ void k_build_pin(const bf16* __restrict__ hb, const bf16* __restrict__ obs_enc,
                            bf16* __restrict__ pin, int t)
{
    int idx = blockIdx.x * blockDim.x + threadIdx.x;
    if (idx >= BB * PIN) return;
    int b = idx / PIN, j = idx - b * PIN;
    pin[idx] = (j < HID) ? hb[b * HID + j]
                         : obs_enc[((size_t)(b * SS + t)) * HID + (j - HID)];
}

__global__ void k_sample_kl(const float* __restrict__ qm, const float* __restrict__ qlv,
                            const float* __restrict__ pm, const float* __restrict__ plv,
                            const float* __restrict__ eps, bf16* __restrict__ zb,
                            float* __restrict__ acc, int t)
{
    __shared__ float red[256];
    int idx = blockIdx.x * 256 + threadIdx.x;     // exactly B*LAT threads
    int b = idx / LAT, j = idx - b * LAT;
    float e  = eps[((size_t)(b * SS + t)) * LAT + j];
    float m  = qm[idx], lv = qlv[idx];
    float z  = m + e * __expf(0.5f * lv);
    zb[idx]  = (bf16)z;
    float dm = m - pm[idx];
    float pl = plv[idx];
    float kl = 0.5f * (pl - lv + (__expf(lv) + dm * dm) * __expf(-pl) - 1.0f);
    red[threadIdx.x] = kl;
    __syncthreads();
    for (int s = 128; s > 0; s >>= 1) {
        if ((int)threadIdx.x < s) red[threadIdx.x] += red[threadIdx.x + s];
        __syncthreads();
    }
    if (threadIdx.x == 0) atomicAdd(acc + 1, red[0]);
}

__global__ void k_build_hz(const bf16* __restrict__ hb, const bf16* __restrict__ zb,
                           bf16* __restrict__ hz)
{
    int idx = blockIdx.x * blockDim.x + threadIdx.x;
    if (idx >= BB * HZD) return;
    int b = idx / HZD, j = idx - b * HZD;
    hz[idx] = (j < HID) ? hb[b * HID + j] : zb[b * LAT + (j - HID)];
}

__global__ void k_recon(const float* __restrict__ pred, const float* __restrict__ outc,
                        float* __restrict__ acc, int t)
{
    __shared__ float red[256];
    int idx = blockIdx.x * 256 + threadIdx.x;     // exactly B*OBS threads
    int b = idx / OBSD, j = idx - b * OBSD;
    float d = pred[idx] - outc[((size_t)(b * SS + t)) * OBSD + j];
    red[threadIdx.x] = d * d;
    __syncthreads();
    for (int s = 128; s > 0; s >>= 1) {
        if ((int)threadIdx.x < s) red[threadIdx.x] += red[threadIdx.x + s];
        __syncthreads();
    }
    if (threadIdx.x == 0) atomicAdd(acc + 0, red[0]);
}

__global__ void k_success(const bf16* __restrict__ sh, const float* __restrict__ Ws2,
                          const float* __restrict__ bs2, const float* __restrict__ suc,
                          float* __restrict__ acc, int t)
{
    int wave = blockIdx.x * ((int)blockDim.x >> 5) + ((int)threadIdx.x >> 5);
    int lane = (int)threadIdx.x & 31;
    if (wave >= BB) return;
    const bf16* row = sh + (size_t)wave * HID;
    float s = 0.0f;
    for (int k = lane; k < HID; k += 32) s += (float)row[k] * Ws2[k];
    for (int m = 16; m > 0; m >>= 1) s += __shfl_xor(s, m, 32);
    if (lane == 0) {
        float ps = sigf(s + bs2[0]);
        ps = fminf(1.0f - 1e-7f, fmaxf(1e-7f, ps));
        float y = suc[(size_t)wave * SS + t];
        float bce = -(y * __logf(ps) + (1.0f - y) * __logf(1.0f - ps));
        atomicAdd(acc + 2, bce);
    }
}

__global__ void k_finalize(const float* __restrict__ acc, float* __restrict__ out)
{
    if (threadIdx.x == 0 && blockIdx.x == 0) {
        float recon = acc[0] / (float)(SS * BB * OBSD);
        float kl    = acc[1] / (float)(SS * BB);
        float bce   = acc[2] / (float)(SS * BB);
        out[0] = recon + kl + bce;
        out[1] = recon;
        out[2] = kl;
        out[3] = bce;
    }
}

// ---------------------------------------------------------------------------

extern "C" void kernel_launch(void* const* d_in, const int* in_sizes, int n_in,
                              void* d_out, int out_size, void* d_ws, size_t ws_size,
                              hipStream_t stream)
{
    (void)in_sizes; (void)n_in; (void)out_size; (void)ws_size;

    // inputs (setup_inputs dict order; params flattened in insertion order)
    const float* obs     = (const float*)d_in[0];
    const int*   actions = (const int*)  d_in[1];
    const float* outc    = (const float*)d_in[2];
    const float* succ    = (const float*)d_in[3];
    const float* eps     = (const float*)d_in[4];
    const float* W_obs = (const float*)d_in[5];  const float* b_obs = (const float*)d_in[6];
    const float* a_tab = (const float*)d_in[7];
    const float* W_ih  = (const float*)d_in[8];  const float* W_hh  = (const float*)d_in[9];
    const float* b_ih  = (const float*)d_in[10]; const float* b_hh  = (const float*)d_in[11];
    const float* W_pr  = (const float*)d_in[12]; const float* b_pr  = (const float*)d_in[13];
    const float* W_prm = (const float*)d_in[14]; const float* b_prm = (const float*)d_in[15];
    const float* W_prv = (const float*)d_in[16]; const float* b_prv = (const float*)d_in[17];
    const float* W_po  = (const float*)d_in[18]; const float* b_po  = (const float*)d_in[19];
    const float* W_pom = (const float*)d_in[20]; const float* b_pom = (const float*)d_in[21];
    const float* W_pov = (const float*)d_in[22]; const float* b_pov = (const float*)d_in[23];
    const float* W_o1  = (const float*)d_in[24]; const float* b_o1  = (const float*)d_in[25];
    const float* W_o2  = (const float*)d_in[26]; const float* b_o2  = (const float*)d_in[27];
    const float* W_s1  = (const float*)d_in[28]; const float* b_s1  = (const float*)d_in[29];
    const float* W_s2  = (const float*)d_in[30]; const float* b_s2  = (const float*)d_in[31];

    // ---- workspace layout (256B-aligned bump allocator) ----
    char* ws = (char*)d_ws;
    size_t off = 0;
    auto alloc = [&](size_t bytes) -> char* {
        char* p = ws + off;
        off = (off + bytes + 255) & ~(size_t)255;
        return p;
    };
    bf16* Wt_obs = (bf16*)alloc((size_t)HID * OBSD * 2);
    bf16* Wt_ih  = (bf16*)alloc((size_t)G3  * GIN  * 2);
    bf16* Wt_hh  = (bf16*)alloc((size_t)G3  * HID  * 2);
    bf16* Wt_pr  = (bf16*)alloc((size_t)HID * HID  * 2);
    bf16* Wt_prm = (bf16*)alloc((size_t)LAT * HID  * 2);
    bf16* Wt_prv = (bf16*)alloc((size_t)LAT * HID  * 2);
    bf16* Wt_po  = (bf16*)alloc((size_t)PIN * PIN  * 2);
    bf16* Wt_pom = (bf16*)alloc((size_t)LAT * PIN  * 2);
    bf16* Wt_pov = (bf16*)alloc((size_t)LAT * PIN  * 2);
    bf16* Wt_o1  = (bf16*)alloc((size_t)HID * HZD  * 2);
    bf16* Wt_o2  = (bf16*)alloc((size_t)OBSD* HID  * 2);
    bf16* Wt_s1  = (bf16*)alloc((size_t)HID * HZD  * 2);
    bf16* obs_enc= (bf16*)alloc((size_t)BB * SS * HID * 2);
    bf16* aemb   = (bf16*)alloc((size_t)BB * SS * AEMB * 2);
    float* h     = (float*)alloc((size_t)BB * HID * 4);
    bf16*  hb    = (bf16*) alloc((size_t)BB * HID * 2);
    bf16*  zb    = (bf16*) alloc((size_t)BB * LAT * 2);
    bf16*  xg    = (bf16*) alloc((size_t)BB * GIN * 2);
    float* gx    = (float*)alloc((size_t)BB * G3 * 4);
    float* gh    = (float*)alloc((size_t)BB * G3 * 4);
    bf16*  hp    = (bf16*) alloc((size_t)BB * HID * 2);
    float* pm    = (float*)alloc((size_t)BB * LAT * 4);
    float* plv   = (float*)alloc((size_t)BB * LAT * 4);
    bf16*  pin   = (bf16*) alloc((size_t)BB * PIN * 2);
    bf16*  hq    = (bf16*) alloc((size_t)BB * PIN * 2);
    float* qm    = (float*)alloc((size_t)BB * LAT * 4);
    float* qlv   = (float*)alloc((size_t)BB * LAT * 4);
    bf16*  hz    = (bf16*) alloc((size_t)BB * HZD * 2);
    bf16*  oh    = (bf16*) alloc((size_t)BB * HID * 2);
    float* pred  = (float*)alloc((size_t)BB * OBSD * 4);
    bf16*  shd   = (bf16*) alloc((size_t)BB * HID * 2);
    float* acc   = (float*)alloc(4 * 4);

    const dim3 blk(256);
    auto grid1  = [](int n) { return dim3((unsigned)((n + 255) / 256)); };
    auto gridG  = [](int M, int N) {
        int waves = (M >> 5) * (N >> 5);            // one wave per 32x32 C block
        return dim3((unsigned)((waves + 7) / 8));   // 8 waves / block
    };

    // ---- setup: zero state + accumulators, convert/transpose weights ----
    k_init<<<grid1(BB * HID), blk, 0, stream>>>(h, hb, zb, acc);
    k_transpose<<<grid1(OBSD * HID), blk, 0, stream>>>(W_obs, Wt_obs, OBSD, HID);
    k_transpose<<<grid1(GIN * G3),   blk, 0, stream>>>(W_ih,  Wt_ih,  GIN,  G3);
    k_transpose<<<grid1(HID * G3),   blk, 0, stream>>>(W_hh,  Wt_hh,  HID,  G3);
    k_transpose<<<grid1(HID * HID),  blk, 0, stream>>>(W_pr,  Wt_pr,  HID,  HID);
    k_transpose<<<grid1(HID * LAT),  blk, 0, stream>>>(W_prm, Wt_prm, HID,  LAT);
    k_transpose<<<grid1(HID * LAT),  blk, 0, stream>>>(W_prv, Wt_prv, HID,  LAT);
    k_transpose<<<grid1(PIN * PIN),  blk, 0, stream>>>(W_po,  Wt_po,  PIN,  PIN);
    k_transpose<<<grid1(PIN * LAT),  blk, 0, stream>>>(W_pom, Wt_pom, PIN,  LAT);
    k_transpose<<<grid1(PIN * LAT),  blk, 0, stream>>>(W_pov, Wt_pov, PIN,  LAT);
    k_transpose<<<grid1(HZD * HID),  blk, 0, stream>>>(W_o1,  Wt_o1,  HZD,  HID);
    k_transpose<<<grid1(HID * OBSD), blk, 0, stream>>>(W_o2,  Wt_o2,  HID,  OBSD);
    k_transpose<<<grid1(HZD * HID),  blk, 0, stream>>>(W_s1,  Wt_s1,  HZD,  HID);
    k_aemb<<<grid1(BB * SS * AEMB), blk, 0, stream>>>(a_tab, actions, aemb);

    // obs_enc = ELU(obs @ W_obs + b_obs) over the whole sequence (M = B*S)
    gemm_wmma<1, true><<<gridG(BB * SS, HID), blk, 0, stream>>>(
        obs, Wt_obs, b_obs, obs_enc, BB * SS, HID, OBSD);

    // ---- sequential scan ----
    for (int t = 0; t < SS; ++t) {
        k_build_x<<<grid1(BB * GIN), blk, 0, stream>>>(zb, aemb, xg, t);
        gemm_wmma<0, false><<<gridG(BB, G3), blk, 0, stream>>>(xg, Wt_ih, b_ih, gx, BB, G3, GIN);
        gemm_wmma<0, false><<<gridG(BB, G3), blk, 0, stream>>>(hb, Wt_hh, b_hh, gh, BB, G3, HID);
        k_gru<<<grid1(BB * HID), blk, 0, stream>>>(gx, gh, h, hb);
        // prior
        gemm_wmma<1, false><<<gridG(BB, HID), blk, 0, stream>>>(hb, Wt_pr, b_pr, hp, BB, HID, HID);
        gemm_wmma<0, false><<<gridG(BB, LAT), blk, 0, stream>>>(hp, Wt_prm, b_prm, pm, BB, LAT, HID);
        gemm_wmma<2, false><<<gridG(BB, LAT), blk, 0, stream>>>(hp, Wt_prv, b_prv, plv, BB, LAT, HID);
        // posterior
        k_build_pin<<<grid1(BB * PIN), blk, 0, stream>>>(hb, obs_enc, pin, t);
        gemm_wmma<1, false><<<gridG(BB, PIN), blk, 0, stream>>>(pin, Wt_po, b_po, hq, BB, PIN, PIN);
        gemm_wmma<0, false><<<gridG(BB, LAT), blk, 0, stream>>>(hq, Wt_pom, b_pom, qm, BB, LAT, PIN);
        gemm_wmma<2, false><<<gridG(BB, LAT), blk, 0, stream>>>(hq, Wt_pov, b_pov, qlv, BB, LAT, PIN);
        // sample + KL
        k_sample_kl<<<dim3(BB * LAT / 256), blk, 0, stream>>>(qm, qlv, pm, plv, eps, zb, acc, t);
        // heads
        k_build_hz<<<grid1(BB * HZD), blk, 0, stream>>>(hb, zb, hz);
        gemm_wmma<1, false><<<gridG(BB, HID), blk, 0, stream>>>(hz, Wt_o1, b_o1, oh, BB, HID, HZD);
        gemm_wmma<0, false><<<gridG(BB, OBSD), blk, 0, stream>>>(oh, Wt_o2, b_o2, pred, BB, OBSD, HID);
        k_recon<<<dim3(BB * OBSD / 256), blk, 0, stream>>>(pred, outc, acc, t);
        gemm_wmma<1, false><<<gridG(BB, HID), blk, 0, stream>>>(hz, Wt_s1, b_s1, shd, BB, HID, HZD);
        k_success<<<dim3(BB / 8), blk, 0, stream>>>(shd, W_s2, b_s2, succ, acc, t);
    }

    k_finalize<<<dim3(1), dim3(64), 0, stream>>>(acc, (float*)d_out);
}
